// SSN_59485297049603
// MI455X (gfx1250) — matrix operand order repaired and checked by reading, not compiled, not run on hardware
//
#include <hip/hip_runtime.h>
#include <hip/hip_bf16.h>

#define HW 512
#define PLANE (HW*HW)

typedef __attribute__((ext_vector_type(16))) _Float16 v16h;
typedef __attribute__((ext_vector_type(8)))  float    v8f;

union V16H { v16h v; _Float16 h[16]; };
union V8F  { v8f  v; float    f[8];  };

__device__ __forceinline__ int iclamp(int v, int lo, int hi) {
    return v < lo ? lo : (v > hi ? hi : v);
}

// ---------------------------------------------------------------------------
// Pack conv weights OIHW f32 -> f16 in B-fragment order:
//   wpack[l][ks][lane][e] = W[ks*32 + (lane>>4)*16 + e][lane&15]
// where W[k][oc], k = ic*9 + ky*3 + kx (matches OIHW flat), zero padded.
// Layer l: 0 -> w0 (3,10), 1..8 -> ws_mid[l-1] (10,10), 9 -> wf (10,5)
// ---------------------------------------------------------------------------
__global__ __launch_bounds__(256)
void SSN_pack_weights(const float* __restrict__ w0,
                      const float* __restrict__ ws_mid,
                      const float* __restrict__ wf,
                      _Float16* __restrict__ wpack)
{
    int idx = blockIdx.x * 256 + threadIdx.x;   // 10 * 1536 = 15360
    if (idx >= 15360) return;
    int l    = idx / 1536;
    int pos  = idx - l * 1536;
    int ks   = pos >> 9;          // 0..2
    int lane = (pos >> 4) & 31;   // 0..31
    int e    = pos & 15;          // 0..15
    int k    = ks * 32 + (lane >> 4) * 16 + e;
    int oc   = lane & 15;
    float v = 0.f;
    if (l == 0) {
        if (k < 27 && oc < 10) v = w0[oc * 27 + k];
    } else if (l <= 8) {
        if (k < 90 && oc < 10) v = ws_mid[(l - 1) * 900 + oc * 90 + k];
    } else {
        if (k < 90 && oc < 5)  v = wf[oc * 90 + k];
    }
    wpack[idx] = (_Float16)v;
}

// ---------------------------------------------------------------------------
// 3x3 SAME conv as implicit GEMM on WMMA.
// Block = 256 threads = 8 waves; block covers 256 contiguous pixels of one
// image row. Each wave computes TWO 16-pixel x 16-ochannel tiles (two
// accumulators sharing each B fragment): 2*NK WMMAs per wave.
// Input patch (CIN x 3 rows x 259 cols, + one zeroed row) staged to LDS as
// f16; invalid K elements read the zero row -> all A loads unconditional.
// Weights staged in B-fragment order (one aligned 32B LDS read per kstep).
// ---------------------------------------------------------------------------
template <int CIN, int NK>
__global__ __launch_bounds__(256)
void SSN_conv3x3_wmma(const float* __restrict__ in,
                      const _Float16* __restrict__ wk,   // [NK][32][16] frags
                      const float* __restrict__ bias, int Cout,
                      float* __restrict__ out, int out_bs_planes,
                      int do_relu)
{
    constexpr int KEFF = CIN * 9;
    constexpr int PC   = 260;                   // padded patch row stride
    constexpr int ZROW = CIN * 3;               // zeroed row index
    __shared__ __align__(32) _Float16 lw[NK * 32 * 16];
    __shared__ _Float16 lpatch[(CIN * 3 + 1) * PC];

    // block position: 256 contiguous pixels within one row of one batch image
    int p0  = blockIdx.x * 256;
    int b   = p0 / PLANE;
    int rem = p0 - b * PLANE;
    int y   = rem >> 9;                         // image row
    int X0  = rem & 511;                        // 256-aligned column base
    const float* inb = in + (long)b * CIN * PLANE;

    // CDNA5 prefetch path (global_prefetch_b8) for the patch rows
    if (threadIdx.x < 3 * CIN) {
        int rr = threadIdx.x, ky = rr % 3, ic = rr / 3;
        int yy = iclamp(y + ky - 1, 0, HW - 1);
        __builtin_prefetch(&inb[(ic * HW + yy) * HW + (X0 > 0 ? X0 - 1 : 0)], 0, 1);
    }

    // stage weights (B-fragment order)
    for (int i = threadIdx.x; i < NK * 512; i += 256) lw[i] = wk[i];

    // stage input patch with SAME zero-padding: rows y-1..y+1, cols X0-1..X0+257
    for (int i = threadIdx.x; i < CIN * 3 * 259; i += 256) {
        int col = i % 259;
        int rr  = i / 259;
        int ky  = rr % 3;
        int ic  = rr / 3;
        int yy  = y + ky - 1;
        int xx  = X0 - 1 + col;
        float v = 0.f;
        if (yy >= 0 && yy < HW && xx >= 0 && xx < HW)
            v = inb[(ic * HW + yy) * HW + xx];
        lpatch[(ic * 3 + ky) * PC + col] = (_Float16)v;
    }
    // zero row for out-of-range K elements
    for (int i = threadIdx.x; i < PC; i += 256)
        lpatch[ZROW * PC + i] = (_Float16)0.f;
    __syncthreads();

    int wave = threadIdx.x >> 5;
    int lane = threadIdx.x & 31;
    int m    = lane & 15;                        // M (pixel) / N (out channel)
    int hi   = lane >> 4;
    int cb0  = wave * 16 + m;                    // tile 0 patch col (kx=0)
    int cb1  = cb0 + 128;                        // tile 1 patch col

    v8f c0 = {}, c1 = {};
    #pragma unroll
    for (int ks = 0; ks < NK; ++ks) {
        const int kb = ks * 32;
        V16H a0, a1, bm;
        // A-fragment: hi=0 elems -> K=kb+{0..15},
        //             hi=1 elems -> K=kb+{16..31}
        #pragma unroll
        for (int e = 0; e < 16; ++e) {
            const int k0  = kb + e;                     // hi=0 K index
            const int k1  = k0 + 16;                    // hi=1 K index
            const int ic0 = k0 / 9, r0 = k0 - ic0 * 9;
            const int ky0 = r0 / 3, kx0 = r0 - ky0 * 3;
            const int ic1 = k1 / 9, r1 = k1 - ic1 * 9;
            const int ky1 = r1 / 3, kx1 = r1 - ky1 * 3;
            const int off0 = (k0 < KEFF) ? (ic0 * 3 + ky0) * PC + kx0
                                         : ZROW * PC;
            const int off1 = (k1 < KEFF) ? (ic1 * 3 + ky1) * PC + kx1
                                         : ZROW * PC;
            int off = hi ? off1 : off0;       // cndmask of two constants
            a0.h[e] = lpatch[off + cb0];      // unconditional LDS loads
            a1.h[e] = lpatch[off + cb1];
        }
        // B-fragment: one aligned 32-byte LDS read per lane, shared by 2 WMMAs
        bm.v = *(const v16h*)&lw[(ks * 32 + lane) * 16];
        c0 = __builtin_amdgcn_wmma_f32_16x16x32_f16(false, a0.v, false, bm.v,
                                                    (short)0, c0, false, false);
        c1 = __builtin_amdgcn_wmma_f32_16x16x32_f16(false, a1.v, false, bm.v,
                                                    (short)0, c1, false, false);
    }

    // D layout: VGPR i -> M = 8*hi + i, N = lane&15
    int oc = m;
    if (oc < Cout) {
        float bb = bias[oc];
        float* o0 = out + ((long)b * out_bs_planes + oc) * PLANE + y * HW
                        + X0 + wave * 16;
        V8F d0, d1; d0.v = c0; d1.v = c1;
        #pragma unroll
        for (int i = 0; i < 8; ++i) {
            int M = hi * 8 + i;
            float v0 = d0.f[i] + bb;
            float v1 = d1.f[i] + bb;
            if (do_relu) { v0 = fmaxf(v0, 0.f); v1 = fmaxf(v1, 0.f); }
            o0[M]       = v0;
            o0[128 + M] = v1;
        }
    }
}

// ---------------------------------------------------------------------------
// imfeat channels 5..7 = X, 8 = Xg (col index), 9 = Yg (row index).
// (channels 0..4 = s1 were written directly by the final conv)
// ---------------------------------------------------------------------------
__global__ __launch_bounds__(256)
void SSN_fill_imfeat(const float* __restrict__ X, float* __restrict__ imfeat)
{
    int idx = blockIdx.x * 256 + threadIdx.x;   // 2*PLANE
    int b = idx / PLANE;
    int r = idx - b * PLANE;
    int y = r >> 9, x = r & 511;
    float* ib = imfeat + (long)b * 10 * PLANE;
    const float* xb = X + (long)b * 3 * PLANE;
    #pragma unroll
    for (int cc = 0; cc < 3; ++cc)
        ib[(5 + cc) * PLANE + r] = xb[cc * PLANE + r];
    ib[8 * PLANE + r] = (float)x;
    ib[9 * PLANE + r] = (float)y;
}

// cent0 = blockmean(imfeat): 32x32 blocks -> (2,10,16,16)
__global__ __launch_bounds__(256)
void SSN_blockmean_feat(const float* __restrict__ imfeat, float* __restrict__ cent)
{
    int idx = blockIdx.x * 256 + threadIdx.x;   // 5120
    if (idx >= 5120) return;
    int bx = idx & 15;
    int by = (idx >> 4) & 15;
    int c  = (idx >> 8) % 10;
    int b  = idx / 2560;
    const float* p = imfeat + ((long)b * 10 + c) * PLANE;
    float s = 0.f;
    for (int ly = 0; ly < 32; ++ly)
        for (int lx = 0; lx < 32; ++lx)
            s += p[(by * 32 + ly) * HW + bx * 32 + lx];
    cent[idx] = s * (1.0f / 1024.0f);
}

// ---------------------------------------------------------------------------
// aff[s] = exp(-mean_c(((f_c - cent_c[clip(yc+dr),clip(xc+dc)])*ck_c)^2)/100)
// ck = 1 for c<8, 10/sqrt(512*512/256)=0.3125 for coord channels.
// Also writes asum = sum_s aff[s].
// ---------------------------------------------------------------------------
__global__ __launch_bounds__(256)
void SSN_affinity(const float* __restrict__ imfeat, const float* __restrict__ cent,
                  float* __restrict__ aff, float* __restrict__ asum)
{
    int idx = blockIdx.x * 256 + threadIdx.x;   // 2*PLANE
    int b = idx / PLANE;
    int r = idx - b * PLANE;
    int y = r >> 9, x = r & 511;
    const float* ib = imfeat + (long)b * 10 * PLANE;
    float f[10];
    #pragma unroll
    for (int c = 0; c < 10; ++c) f[c] = ib[c * PLANE + r];
    int yc = y >> 5, xc = x >> 5;
    const float* cb = cent + (long)b * 2560;
    float sa = 0.f;
    #pragma unroll
    for (int s = 0; s < 9; ++s) {
        int dr = s / 3 - 1, dc = s % 3 - 1;
        int y1 = iclamp(yc + dr, 0, 15);
        int x1 = iclamp(xc + dc, 0, 15);
        float acc = 0.f;
        #pragma unroll
        for (int c = 0; c < 10; ++c) {
            float ckc = (c < 8) ? 1.0f : 0.3125f;
            float d = (f[c] - cb[(c * 16 + y1) * 16 + x1]) * ckc;
            acc += d * d;
        }
        float a = __expf(-acc * (1.0f / 1000.0f));   // mean over 10 ch, /CONST=100
        aff[(long)(b * 9 + s) * PLANE + r] = a;
        sa += a;
    }
    asum[(long)b * PLANE + r] = sa;
}

// Per 32x32 block: bm_feat[c] = mean(imfeat_c * asum), bm_asum = mean(asum)
__global__ __launch_bounds__(256)
void SSN_blocksum(const float* __restrict__ imfeat, const float* __restrict__ asum,
                  float* __restrict__ bm_feat, float* __restrict__ bm_asum)
{
    __shared__ float red[256];
    int blk = blockIdx.x;           // 512: b*256 + by*16 + bx
    int b  = blk >> 8;
    int by = (blk >> 4) & 15;
    int bx = blk & 15;
    int tid = threadIdx.x;
    float loc[11];
    #pragma unroll
    for (int q = 0; q < 11; ++q) loc[q] = 0.f;
    const float* ib = imfeat + (long)b * 10 * PLANE;
    const float* ab = asum + (long)b * PLANE;
    for (int p = tid; p < 1024; p += 256) {
        int ly = p >> 5, lx = p & 31;
        int off = (by * 32 + ly) * HW + bx * 32 + lx;
        float a = ab[off];
        loc[10] += a;
        #pragma unroll
        for (int c = 0; c < 10; ++c) loc[c] += ib[c * PLANE + off] * a;
    }
    #pragma unroll
    for (int q = 0; q < 11; ++q) {
        red[tid] = loc[q];
        __syncthreads();
        for (int s = 128; s > 0; s >>= 1) {
            if (tid < s) red[tid] += red[tid + s];
            __syncthreads();
        }
        if (tid == 0) {
            float mean = red[0] * (1.0f / 1024.0f);
            if (q < 10) bm_feat[((b * 10 + q) * 16 + by) * 16 + bx] = mean;
            else        bm_asum[(b * 16 + by) * 16 + bx] = mean;
        }
        __syncthreads();
    }
}

// cent = box3(bm_feat) / box3(bm_asum), 3x3 ones SAME on 16x16 grid
__global__ __launch_bounds__(256)
void SSN_box3div(const float* __restrict__ bm_feat, const float* __restrict__ bm_asum,
                 float* __restrict__ cent)
{
    int idx = blockIdx.x * 256 + threadIdx.x;   // 5120
    if (idx >= 5120) return;
    int bx = idx & 15;
    int by = (idx >> 4) & 15;
    int c  = (idx >> 8) % 10;
    int b  = idx / 2560;
    float num = 0.f, den = 0.f;
    for (int di = -1; di <= 1; ++di)
        for (int dj = -1; dj <= 1; ++dj) {
            int i = by + di, j = bx + dj;
            if (i >= 0 && i < 16 && j >= 0 && j < 16) {
                num += bm_feat[((b * 10 + c) * 16 + i) * 16 + j];
                den += bm_asum[(b * 16 + i) * 16 + j];
            }
        }
    cent[idx] = num / den;
}

// ---------------------------------------------------------------------------
extern "C" void kernel_launch(void* const* d_in, const int* in_sizes, int n_in,
                              void* d_out, int out_size, void* d_ws, size_t ws_size,
                              hipStream_t stream)
{
    const float* X      = (const float*)d_in[0];
    const float* w0     = (const float*)d_in[1];
    const float* b0     = (const float*)d_in[2];
    const float* ws_mid = (const float*)d_in[3];
    const float* bs_mid = (const float*)d_in[4];
    const float* wf     = (const float*)d_in[5];
    const float* bf     = (const float*)d_in[6];

    // workspace layout (floats): bufA | bufB | imfeat | small tail
    float* bufA   = (float*)d_ws;                 // 20 planes (2 batches x 10ch)
    float* bufB   = bufA + (size_t)20 * PLANE;
    float* imfeat = bufB + (size_t)20 * PLANE;
    char*  tail   = (char*)(imfeat + (size_t)20 * PLANE);
    _Float16* wpack = (_Float16*)tail;            // 10 * 1536 halves
    float* centA = (float*)(tail + 15360 * sizeof(_Float16));
    float* centB = centA + 5120;
    float* bmf   = centB + 5120;                  // 5120
    float* bma   = bmf + 5120;                    // 512

    float* outp = (float*)d_out;
    float* aff  = outp + 5120;                    // final aff written in place
    float* asum = bufB;                           // reuse after convs (stream-ordered)

    SSN_pack_weights<<<60, 256, 0, stream>>>(w0, ws_mid, wf, wpack);

    // conv0: X(3ch) -> bufA, relu   (2048 blocks x 256 px)
    SSN_conv3x3_wmma<3, 1><<<2048, 256, 0, stream>>>(X, wpack, b0, 10,
                                                     bufA, 10, 1);
    // 8 mid layers, ping-pong
    float* pin = bufA; float* pout = bufB;
    for (int l = 1; l <= 8; ++l) {
        SSN_conv3x3_wmma<10, 3><<<2048, 256, 0, stream>>>(pin, wpack + l * 1536,
                                                          bs_mid + (l - 1) * 10, 10,
                                                          pout, 10, 1);
        float* t = pin; pin = pout; pout = t;
    }
    // final conv: -> imfeat channels 0..4, no relu (batch stride = 10 planes)
    SSN_conv3x3_wmma<10, 3><<<2048, 256, 0, stream>>>(pin, wpack + 9 * 1536,
                                                      bf, 5, imfeat, 10, 0);

    SSN_fill_imfeat<<<2048, 256, 0, stream>>>(X, imfeat);
    SSN_blockmean_feat<<<20, 256, 0, stream>>>(imfeat, centA);

    // 10 SSN iterations: aff = affinity(cent); cent = centroids(aff)
    float* cent = centA; float* centN = centB;
    for (int it = 0; it < 10; ++it) {
        SSN_affinity<<<2048, 256, 0, stream>>>(imfeat, cent, aff, asum);
        SSN_blocksum<<<512, 256, 0, stream>>>(imfeat, asum, bmf, bma);
        SSN_box3div<<<20, 256, 0, stream>>>(bmf, bma, centN);
        float* t = cent; cent = centN; centN = t;
    }

    // outputs: cent (5120 floats) then aff (already in place)
    hipMemcpyAsync(outp, cent, 5120 * sizeof(float),
                   hipMemcpyDeviceToDevice, stream);
}
